// MicroMoE_2637109920535
// MI455X (gfx1250) — compile-verified, matrix operand order checked by reading, not run on hardware
//
#include <hip/hip_runtime.h>
#include <hip/hip_bf16.h>
#include <stdint.h>

// ---------------------------------------------------------------------------
// MicroMoE on MI455X (gfx1250): top-2 gathered expert FFN on bf16 WMMA.
//   B=4 T=4096 H=1024  E=8 K=2 F=1024   N = 16384 tokens
// d_out = [N*H floats for out] + [1 float lb_loss]
// Workspace use: ~49 MB (bf16 weights + routing metadata).
// ---------------------------------------------------------------------------

#define NTOK   16384
#define HDIM   1024
#define FDIM   1024
#define NEXP   8
#define NPAIR  (NTOK * 2)        // 32768 (token, expert) assignments
#define MAXPAD 33792             // padded pair slots upper bound (32768 + 8*63, rounded)
#define MAXTIL 528               // max 64-token tiles (<= 520 actually)
#define LDW    1032              // LDS row stride in bf16 elements (pad vs bank conflicts)
#define WELEM  (NEXP * FDIM * HDIM)   // 8,388,608 elements per weight tensor

typedef __bf16 bhalf;
typedef bhalf v16bf __attribute__((ext_vector_type(16)));
typedef float v8f  __attribute__((ext_vector_type(8)));

__device__ __forceinline__ unsigned short f2bf(float f) {
  union { float f; unsigned u; } a; a.f = f;
  unsigned u = a.u;
  u += 0x7FFFu + ((u >> 16) & 1u);          // round-to-nearest-even
  return (unsigned short)(u >> 16);
}

// 16-bit A/B fragment load (ISA 7.12.2 layout): lane&15 = row (M or N),
// lane>>4 selects which two 8-element contiguous K-chunks this lane holds:
//   half 0: K = [c0 .. c0+7]  and [c0+16 .. c0+23]
//   half 1: K = [c0+8 .. c0+15] and [c0+24 .. c0+31]
// Two aligned 16-byte loads (ds_load_b128 / global_load_b128).
__device__ __forceinline__ v16bf load_frag(const unsigned short* p0, int ld,
                                           int r0, int c0, int lane) {
  const unsigned short* p =
      p0 + (size_t)(r0 + (lane & 15)) * (size_t)ld + (size_t)(c0 + ((lane >> 4) << 3));
  union { uint4 q[2]; v16bf v; } u;
  u.q[0] = *(const uint4*)(p);
  u.q[1] = *(const uint4*)(p + 16);
  return u.v;
}

__device__ __forceinline__ v8f wmma_bf16(v16bf a, v16bf b, v8f c) {
  // (neg_a, A, neg_b, B, c_mod, C, reuse_a, reuse_b)
  return __builtin_amdgcn_wmma_f32_16x16x32_bf16(false, a, false, b, (short)0, c,
                                                 false, false);
}

// ---------------------------------------------------------------- init -----
__global__ void moe_init(float* __restrict__ out, size_t n, unsigned* __restrict__ ctrl) {
  size_t i = (size_t)blockIdx.x * blockDim.x + threadIdx.x;
  size_t stride = (size_t)gridDim.x * blockDim.x;
  for (size_t j = i; j < n; j += stride) out[j] = 0.0f;
  if (i < 48) ctrl[i] = 0u;   // usage[16 pad] counts[16 pad] fill[16 pad]
}

// ------------------------------------------------- fp32 -> bf16 weights ----
__global__ void moe_cvt(const float* __restrict__ g, const float* __restrict__ u,
                        const float* __restrict__ d,
                        unsigned short* __restrict__ og, unsigned short* __restrict__ ou,
                        unsigned short* __restrict__ od) {
  size_t i = (size_t)blockIdx.x * blockDim.x + threadIdx.x;
  if (i < (size_t)WELEM) {
    og[i] = f2bf(g[i]);
    ou[i] = f2bf(u[i]);
    od[i] = f2bf(d[i]);
  }
}

// ------------------------------------------------------------- router ------
// One wave32 per token: 8 dot-products over H=1024, wave reduce, softmax,
// top-2 (stable order), normalized gates; usage + count atomics.
__global__ __launch_bounds__(256) void moe_router(
    const float* __restrict__ x, const float* __restrict__ rw,
    int* __restrict__ topi, float* __restrict__ topg,
    float* __restrict__ usage, unsigned* __restrict__ counts) {
  __shared__ float sRW[NEXP * HDIM];        // 32 KB
  for (int i = threadIdx.x; i < NEXP * HDIM; i += 256) sRW[i] = rw[i];
  __syncthreads();

  int wave = threadIdx.x >> 5, lane = threadIdx.x & 31;
  int t = blockIdx.x * 8 + wave;
  const float* xr = x + (size_t)t * HDIM;

  float acc[NEXP];
#pragma unroll
  for (int e = 0; e < NEXP; ++e) acc[e] = 0.0f;

  for (int h = lane; h < HDIM; h += 32) {
    float xv = xr[h];
#pragma unroll
    for (int e = 0; e < NEXP; ++e) acc[e] += xv * sRW[e * HDIM + h];
  }
#pragma unroll
  for (int e = 0; e < NEXP; ++e) {
#pragma unroll
    for (int m = 16; m >= 1; m >>= 1) acc[e] += __shfl_xor(acc[e], m, 32);
  }

  if (lane == 0) {
    float mx = acc[0];
#pragma unroll
    for (int e = 1; e < NEXP; ++e) mx = fmaxf(mx, acc[e]);
    float p[NEXP], s = 0.0f;
#pragma unroll
    for (int e = 0; e < NEXP; ++e) { p[e] = __expf(acc[e] - mx); s += p[e]; }
    float inv = 1.0f / s;
#pragma unroll
    for (int e = 0; e < NEXP; ++e) p[e] *= inv;

    int i0 = 0;
#pragma unroll
    for (int e = 1; e < NEXP; ++e) if (p[e] > p[i0]) i0 = e;
    int i1 = (i0 == 0) ? 1 : 0;
#pragma unroll
    for (int e = 0; e < NEXP; ++e) if (e != i0 && p[e] > p[i1]) i1 = e;

    float den = p[i0] + p[i1];
    topi[t * 2 + 0] = i0;  topg[t * 2 + 0] = p[i0] / den;
    topi[t * 2 + 1] = i1;  topg[t * 2 + 1] = p[i1] / den;
#pragma unroll
    for (int e = 0; e < NEXP; ++e) atomicAdd(&usage[e], p[e]);
    atomicAdd(&counts[i0], 1u);
    atomicAdd(&counts[i1], 1u);
  }
}

// ------------------------------------------------------------- lb loss -----
__global__ void moe_lb(const float* __restrict__ usage, float* __restrict__ out_scalar) {
  if (threadIdx.x == 0 && blockIdx.x == 0) {
    float s = 0.0f;
    for (int e = 0; e < NEXP; ++e) {
      float m = usage[e] * (1.0f / (float)NTOK);
      s += m * m;
    }
    *out_scalar = 0.01f * (float)NEXP * s;
  }
}

// ------------------------------------------------------------- scan --------
// Pad each expert segment to a multiple of 64 tokens; build tile map.
__global__ void moe_scan(const unsigned* __restrict__ counts, unsigned* __restrict__ basep,
                         int* __restrict__ ptok, float* __restrict__ pgate,
                         int* __restrict__ ntiles, int* __restrict__ tile_e,
                         int* __restrict__ tile_s) {
  __shared__ int sTot;
  if (threadIdx.x == 0) {
    unsigned tot = 0; int nt = 0;
    for (int e = 0; e < NEXP; ++e) {
      basep[e] = tot;
      unsigned padded = (counts[e] + 63u) & ~63u;
      unsigned tcnt = padded >> 6;
      for (unsigned i = 0; i < tcnt; ++i) {
        tile_e[nt] = e;
        tile_s[nt] = (int)(tot + i * 64u);
        ++nt;
      }
      tot += padded;
    }
    *ntiles = nt;
    sTot = (int)tot;
  }
  __syncthreads();
  int tot = sTot;
  for (int i = threadIdx.x; i < tot; i += blockDim.x) {
    ptok[i] = -1;
    pgate[i] = 0.0f;
  }
}

// ------------------------------------------------------------- gather ------
__global__ void moe_gather(const int* __restrict__ topi, const float* __restrict__ topg,
                           const unsigned* __restrict__ basep, unsigned* __restrict__ fill,
                           int* __restrict__ ptok, float* __restrict__ pgate) {
  int i = blockIdx.x * blockDim.x + threadIdx.x;
  if (i >= NPAIR) return;
  int e = topi[i];
  unsigned pos = atomicAdd(&fill[e], 1u);
  unsigned slot = basep[e] + pos;
  ptok[slot] = i >> 1;
  pgate[slot] = topg[i];
}

// ------------------------------------------------------------- expert FFN --
// One workgroup per 64-token tile of one expert.
// Phase 1: hid = silu(X Gw^T) * (X Uw^T)   -> LDS (bf16)
// Phase 2: out += gate * (hid Dw^T)        -> global_atomic_add_f32
__global__ __launch_bounds__(256) void moe_ffn(
    const float* __restrict__ x,
    const unsigned short* __restrict__ wg, const unsigned short* __restrict__ wu,
    const unsigned short* __restrict__ wd,
    const int* __restrict__ ntiles, const int* __restrict__ tile_e,
    const int* __restrict__ tile_s,
    const int* __restrict__ ptok, const float* __restrict__ pgate,
    float* __restrict__ out) {
  int nt = *ntiles;
  if ((int)blockIdx.x >= nt) return;

  __shared__ __align__(16) unsigned short sX[64 * LDW];   // 129 KB bf16 x-tile
  __shared__ __align__(16) unsigned short sH[64 * LDW];   // 129 KB bf16 hid-tile
  __shared__ float sG[64];
  __shared__ int   sT[64];

  int e = tile_e[blockIdx.x];
  int start = tile_s[blockIdx.x];
  if (threadIdx.x < 64) {
    sT[threadIdx.x] = ptok[start + threadIdx.x];
    sG[threadIdx.x] = pgate[start + threadIdx.x];
  }
  __syncthreads();

  // Stage x-tile as bf16 (padded rows -> zero).
  for (int i = threadIdx.x; i < 64 * HDIM; i += 256) {
    int r = i >> 10, c = i & (HDIM - 1);
    int tok = sT[r];
    float v = (tok >= 0) ? x[(size_t)tok * HDIM + c] : 0.0f;
    sX[r * LDW + c] = f2bf(v);
  }
  __syncthreads();

  int wave = threadIdx.x >> 5, lane = threadIdx.x & 31;
  const unsigned short* gbase = wg + (size_t)e * WELEM / NEXP * NEXP / NEXP * 0 + (size_t)e * (size_t)FDIM * HDIM;
  const unsigned short* ubase = wu + (size_t)e * (size_t)FDIM * HDIM;
  const unsigned short* dbase = wd + (size_t)e * (size_t)HDIM * FDIM;

  // ---- Phase 1: 4 M-tiles x 64 F-tiles of 16x16, K = 1024 in steps of 32.
  for (int tile = wave; tile < 4 * 64; tile += 8) {
    int mt = tile & 3, ft = tile >> 2;
    v8f ag = {}; v8f au = {};
    for (int k = 0; k < HDIM; k += 32) {
      if (k + 64 < HDIM)
        __builtin_prefetch(gbase + (size_t)(ft * 16 + (lane & 15)) * HDIM + k + 64, 0, 1);
      v16bf a  = load_frag(sX, LDW, mt * 16, k, lane);
      v16bf bg = load_frag(gbase, HDIM, ft * 16, k, lane);  // row f of Gw = col of B
      v16bf bu = load_frag(ubase, HDIM, ft * 16, k, lane);
      ag = wmma_bf16(a, bg, ag);
      au = wmma_bf16(a, bu, au);
    }
    int n  = ft * 16 + (lane & 15);
    int mb = mt * 16 + ((lane >> 4) << 3);     // C/D layout: VGPR r -> row r (+8 hi half)
#pragma unroll
    for (int r = 0; r < 8; ++r) {
      float gv = ag[r];
      float hv = (gv / (1.0f + __expf(-gv))) * au[r];      // silu(g) * u
      sH[(mb + r) * LDW + n] = f2bf(hv);
    }
  }
  __syncthreads();

  // ---- Phase 2: out tile [64, 1024] = hid @ Dw^T, gate-scaled scatter-add.
  for (int tile = wave; tile < 4 * 64; tile += 8) {
    int mt = tile & 3, ht = tile >> 2;
    v8f acc = {};
    for (int k = 0; k < FDIM; k += 32) {
      if (k + 64 < FDIM)
        __builtin_prefetch(dbase + (size_t)(ht * 16 + (lane & 15)) * FDIM + k + 64, 0, 1);
      v16bf a = load_frag(sH, LDW, mt * 16, k, lane);
      v16bf b = load_frag(dbase, FDIM, ht * 16, k, lane);
      acc = wmma_bf16(a, b, acc);
    }
    int n  = ht * 16 + (lane & 15);
    int mb = mt * 16 + ((lane >> 4) << 3);
#pragma unroll
    for (int r = 0; r < 8; ++r) {
      int m = mb + r;
      int tok = sT[m];
      if (tok >= 0) atomicAdd(&out[(size_t)tok * HDIM + n], sG[m] * acc[r]);
    }
  }
}

// ---------------------------------------------------------------------------
extern "C" void kernel_launch(void* const* d_in, const int* in_sizes, int n_in,
                              void* d_out, int out_size, void* d_ws, size_t ws_size,
                              hipStream_t stream) {
  const float* x  = (const float*)d_in[0];   // [N, H]
  const float* rw = (const float*)d_in[1];   // [E, H]
  const float* gw = (const float*)d_in[2];   // [E, F, H]
  const float* uw = (const float*)d_in[3];   // [E, F, H]
  const float* dw = (const float*)d_in[4];   // [E, H, F]
  float* out = (float*)d_out;                // [N*H] + [1] lb_loss

  // Workspace carve (~48.8 MB total; ws_size assumed >= 64 MB).
  char* ws = (char*)d_ws;
  unsigned short* wg_bf = (unsigned short*)ws;  ws += (size_t)WELEM * 2;
  unsigned short* wu_bf = (unsigned short*)ws;  ws += (size_t)WELEM * 2;
  unsigned short* wd_bf = (unsigned short*)ws;  ws += (size_t)WELEM * 2;
  float*    usage  = (float*)ws;     ws += 64;      // 8 used, 64B padded
  unsigned* counts = (unsigned*)ws;  ws += 64;
  unsigned* fill   = (unsigned*)ws;  ws += 64;
  unsigned* basep  = (unsigned*)ws;  ws += 64;
  int*      ntiles = (int*)ws;       ws += 64;
  int*      tile_e = (int*)ws;       ws += MAXTIL * 4;
  int*      tile_s = (int*)ws;       ws += MAXTIL * 4;
  int*      topi   = (int*)ws;       ws += NPAIR * 4;
  float*    topg   = (float*)ws;     ws += NPAIR * 4;
  int*      ptok   = (int*)ws;       ws += MAXPAD * 4;
  float*    pgate  = (float*)ws;     ws += MAXPAD * 4;

  moe_init<<<8192, 256, 0, stream>>>(out, (size_t)NTOK * HDIM + 1, (unsigned*)usage);
  moe_cvt<<<(WELEM + 255) / 256, 256, 0, stream>>>(gw, uw, dw, wg_bf, wu_bf, wd_bf);
  moe_router<<<NTOK / 8, 256, 0, stream>>>(x, rw, topi, topg, usage, counts);
  moe_lb<<<1, 32, 0, stream>>>(usage, out + (size_t)NTOK * HDIM);
  moe_scan<<<1, 256, 0, stream>>>(counts, basep, ptok, pgate, ntiles, tile_e, tile_s);
  moe_gather<<<NPAIR / 256, 256, 0, stream>>>(topi, topg, basep, fill, ptok, pgate);
  moe_ffn<<<MAXTIL, 256, 0, stream>>>(x, wg_bf, wu_bf, wd_bf, ntiles, tile_e, tile_s,
                                      ptok, pgate, out);
}
